// BasicCae_39745627357481
// MI455X (gfx1250) — matrix operand order repaired and checked by reading, not compiled, not run on hardware
//
#include <hip/hip_runtime.h>

#define BATCH 256
#define IN    28224          // input size per sample
#define FEAT  1500           // real feature dim
#define FEATP 1536           // padded feature dim (24 * 64)
#define BK    64             // K slice per stage
#define LDT   72             // LDS row stride in bf16 elems (BK + 8 pad, 144B, 16B aligned)
#define TILE  (64 * LDT)     // one LDS tile in bf16 elems

#if __has_builtin(__builtin_amdgcn_global_load_async_to_lds_b128) && \
    __has_builtin(__builtin_amdgcn_s_wait_asynccnt)
#define CAE_ASYNC 1
#define NBUF 3
#else
#define CAE_ASYNC 0
#define NBUF 2
#endif

typedef __attribute__((ext_vector_type(16))) __bf16 v16bf;
typedef __attribute__((ext_vector_type(8)))  __bf16 v8bf;
typedef __attribute__((ext_vector_type(2)))  __bf16 v2bf;
typedef __attribute__((ext_vector_type(8)))  float  v8f;
typedef __attribute__((ext_vector_type(4)))  int    v4i;

union Frag16 { v16bf v; v8bf h[2]; };

// f32 pair -> packed bf16 (round-half-up: +0x8000 then take high halves via v_perm_b32)
__device__ __forceinline__ unsigned int pack2_bf16(float a, float b) {
#if __has_builtin(__builtin_amdgcn_cvt_pk_bf16_f32)
  v2bf r = __builtin_amdgcn_cvt_pk_bf16_f32(a, b);
  return __builtin_bit_cast(unsigned int, r);
#else
  unsigned int ua = __float_as_uint(a) + 0x8000u;
  unsigned int ub = __float_as_uint(b) + 0x8000u;
  return __builtin_amdgcn_perm(ub, ua, 0x07060302u);
#endif
}
__device__ __forceinline__ unsigned short f32_bf16(float a) {
  return (unsigned short)(pack2_bf16(a, 0.f) & 0xFFFFu);
}
__device__ __forceinline__ float sigmoidf(float c) {
  return 1.0f / (1.0f + __expf(-c));
}
__device__ __forceinline__ uint4 cvt4x2(float4 a, float4 b) {
  uint4 u;
  u.x = pack2_bf16(a.x, a.y);
  u.y = pack2_bf16(a.z, a.w);
  u.z = pack2_bf16(b.x, b.y);
  u.w = pack2_bf16(b.z, b.w);
  return u;
}

#if CAE_ASYNC
typedef __attribute__((address_space(1))) v4i glb_v4i;
typedef __attribute__((address_space(3))) v4i lds_v4i;
__device__ __forceinline__ void async_b128(const unsigned short* g, unsigned short* l) {
  __builtin_amdgcn_global_load_async_to_lds_b128((glb_v4i*)g, (lds_v4i*)l, 0, 0);
}
#endif

// ---------------- x (f32) -> bf16, one pass ----------------
__global__ __launch_bounds__(256) void cae_cvt_x(const float* __restrict__ x,
                                                 unsigned short* __restrict__ xb) {
  const size_t i = (size_t)(blockIdx.x * 256 + threadIdx.x) * 8;
  float4 a = *(const float4*)(x + i);
  float4 b = *(const float4*)(x + i + 4);
  *(uint4*)(xb + i) = cvt4x2(a, b);
}

// ------- W_enc: row_norm2 + bf16 copy (single read of W_enc) -------
__global__ __launch_bounds__(256) void cae_prep_wenc(const float* __restrict__ Wenc,
                                                     float* __restrict__ rn2,
                                                     unsigned short* __restrict__ wb) {
  const size_t base = (size_t)blockIdx.x * IN;
  const float4* row = (const float4*)(Wenc + base);
  unsigned short* orow = wb + base;
  float s = 0.f;
  for (int c = threadIdx.x; c < IN / 8; c += 256) {
    float4 a = row[2 * c];
    float4 b = row[2 * c + 1];
    *(uint4*)(orow + (size_t)c * 8) = cvt4x2(a, b);
    s += a.x * a.x + a.y * a.y + a.z * a.z + a.w * a.w;
    s += b.x * b.x + b.y * b.y + b.z * b.z + b.w * b.w;
  }
  for (int off = 16; off; off >>= 1) s += __shfl_xor(s, off, 32);
  __shared__ float red[8];
  if ((threadIdx.x & 31) == 0) red[threadIdx.x >> 5] = s;
  __syncthreads();
  if (threadIdx.x == 0) {
    float t = 0.f;
    for (int i = 0; i < 8; ++i) t += red[i];
    rn2[blockIdx.x] = t;
  }
}

// ------- W_dec [28224][1500] f32 -> bf16 [28224][1536], zero-padded K -------
__global__ __launch_bounds__(192) void cae_cvt_wdec(const float* __restrict__ Wdec,
                                                    unsigned short* __restrict__ wb) {
  const int r = blockIdx.x;            // 0..28223
  const int k = threadIdx.x * 8;       // 0..1528
  const float* src = Wdec + (size_t)r * FEAT;
  const float4 z4 = {0.f, 0.f, 0.f, 0.f};
  float4 a = (k + 3 < FEAT) ? *(const float4*)(src + k) : z4;
  float4 b = (k + 7 < FEAT) ? *(const float4*)(src + k + 4) : z4;
  *(uint4*)(wb + (size_t)r * FEATP + k) = cvt4x2(a, b);
}

// ---------------- shared GEMM main loop (all-bf16 operands) -----------------
// Block 64x64x64; 8 waves; wave tile 32x16 (2 accumulators).
// Async path: 3 LDS buffers, 2 tiles in flight via GLOBAL_LOAD_ASYNC_TO_LDS;
// last two iterations peeled so the steady-state loop has no conditionals.
// Fallback: 2 LDS buffers, register-staged global->LDS pipeline.
template <int NSTEP>
__device__ __forceinline__ void gemm_mainloop(
    const unsigned short* __restrict__ pA0, const unsigned short* __restrict__ pA1,
    const unsigned short* __restrict__ pB0, const unsigned short* __restrict__ pB1,
    unsigned short* As, unsigned short* Bs, int srow, int scol,
    const unsigned short* bfragp, const unsigned short* afragp0,
    const unsigned short* afragp1, v8f& acc0, v8f& acc1) {
  const int sofs = srow * LDT + scol;

  auto compute = [&](int cur) {
    const int par = cur * TILE;
#pragma unroll
    for (int ks = 0; ks < 2; ++ks) {
      const int ko = par + ks * 32;
      Frag16 bf, a0, a1;
      bf.h[0] = *(const v8bf*)(bfragp + ko);
      bf.h[1] = *(const v8bf*)(bfragp + ko + 16);
      a0.h[0] = *(const v8bf*)(afragp0 + ko);
      a0.h[1] = *(const v8bf*)(afragp0 + ko + 16);
      a1.h[0] = *(const v8bf*)(afragp1 + ko);
      a1.h[1] = *(const v8bf*)(afragp1 + ko + 16);
      acc0 = __builtin_amdgcn_wmma_f32_16x16x32_bf16(false, a0.v, false, bf.v,
                                                     (short)0, acc0, false, false);
      acc1 = __builtin_amdgcn_wmma_f32_16x16x32_bf16(false, a1.v, false, bf.v,
                                                     (short)0, acc1, false, false);
    }
  };

#if CAE_ASYNC
  auto issue = [&](int tile, int nb) {
    const int o = tile * BK;
    const int bofs = nb * TILE + sofs;
    async_b128(pA0 + o, As + bofs);
    async_b128(pA1 + o, As + bofs + 32 * LDT);
    async_b128(pB0 + o, Bs + bofs);
    async_b128(pB1 + o, Bs + bofs + 32 * LDT);
  };
  // prologue: tiles 0 and 1 in flight
  issue(0, 0);
  issue(1, 1);
  int cur = 0;
#pragma unroll 1
  for (int s = 0; s < NSTEP - 2; ++s) {
    __builtin_amdgcn_s_wait_asynccnt(4);   // oldest tile landed (in-order)
    __syncthreads();
    const int nb = (cur == 0) ? 2 : cur - 1;  // (cur+2)%3
    issue(s + 2, nb);
    compute(cur);
    cur = (cur == 2) ? 0 : cur + 1;
  }
  // tail step NSTEP-2 (two tiles possibly outstanding)
  __builtin_amdgcn_s_wait_asynccnt(4);
  __syncthreads();
  compute(cur);
  cur = (cur == 2) ? 0 : cur + 1;
  // tail step NSTEP-1 (last tile)
  __builtin_amdgcn_s_wait_asynccnt(0);
  __syncthreads();
  compute(cur);
#else
  uint4 rA0 = *(const uint4*)pA0;
  uint4 rA1 = *(const uint4*)pA1;
  uint4 rB0 = *(const uint4*)pB0;
  uint4 rB1 = *(const uint4*)pB1;
#pragma unroll 1
  for (int s = 0; s < NSTEP; ++s) {
    const int par = (s & 1) * TILE;
    *(uint4*)&As[par + sofs]            = rA0;
    *(uint4*)&As[par + sofs + 32 * LDT] = rA1;
    *(uint4*)&Bs[par + sofs]            = rB0;
    *(uint4*)&Bs[par + sofs + 32 * LDT] = rB1;
    __syncthreads();
    if (s + 1 < NSTEP) {
      const int o = (s + 1) * BK;
      rA0 = *(const uint4*)(pA0 + o);
      rA1 = *(const uint4*)(pA1 + o);
      rB0 = *(const uint4*)(pB0 + o);
      rB1 = *(const uint4*)(pB1 + o);
      if (s + 3 < NSTEP) {
        __builtin_prefetch(pA0 + (size_t)(s + 3) * BK, 0, 3);
        __builtin_prefetch(pB0 + (size_t)(s + 3) * BK, 0, 3);
      }
    }
    compute(s & 1);
  }
#endif
}

// ---------------- encoder GEMM: y_enc = sigmoid(xb @ wencb^T + b) ----------
__global__ __launch_bounds__(256) void cae_encoder(
    const unsigned short* __restrict__ xb, const unsigned short* __restrict__ wb,
    const float* __restrict__ benc, const float* __restrict__ rn2,
    unsigned short* __restrict__ yenc, float* __restrict__ jacPart) {
  __shared__ __align__(16) unsigned short As[NBUF * TILE];
  __shared__ __align__(16) unsigned short Bs[NBUF * TILE];
  __shared__ float red[8];

  const int t = threadIdx.x;
  const int blockM = blockIdx.y * 64;
  const int blockN = blockIdx.x * 64;

  const int srow = t >> 3;        // 0..31
  const int scol = (t & 7) * 8;   // 0..56
  const unsigned short* pA0 = xb + (size_t)(blockM + srow) * IN + scol;
  const unsigned short* pA1 = pA0 + (size_t)32 * IN;
  // B rows >= FEAT: clamp to row 0 (finite garbage); those C columns are
  // masked in the epilogue, so no zero-fill is needed (keeps loads plain).
  const int gbn0 = blockN + srow, gbn1 = gbn0 + 32;
  const unsigned short* pB0 = wb + (size_t)(gbn0 < FEAT ? gbn0 : 0) * IN + scol;
  const unsigned short* pB1 = wb + (size_t)(gbn1 < FEAT ? gbn1 : 0) * IN + scol;

  const int lane = t & 31;
  const int w    = t >> 5;
  const int wm   = w >> 2;       // 0..1
  const int wn   = w & 3;        // 0..3
  const int lrow = lane & 15;
  const int kg   = lane >> 4;

  const unsigned short* bfragp  = Bs + (wn * 16 + lrow) * LDT + kg * 8;
  const unsigned short* afragp0 = As + (wm * 32 + lrow) * LDT + kg * 8;
  const unsigned short* afragp1 = afragp0 + 16 * LDT;

  v8f acc0 = {};
  v8f acc1 = {};
  gemm_mainloop<IN / BK>(pA0, pA1, pB0, pB1, As, Bs, srow, scol, bfragp, afragp0,
                         afragp1, acc0, acc1);

  // epilogue: sigmoid, store bf16 y_enc (zero pad cols), jac partial
  const int gn = blockN + wn * 16 + lrow;
  const bool nok = gn < FEAT;
  const float rnv = nok ? rn2[gn] : 0.f;
  const float bev = nok ? benc[gn] : 0.f;
  const int mo = kg * 8;
  float jac = 0.f;
#pragma unroll
  for (int mt = 0; mt < 2; ++mt) {
    const v8f acc = mt ? acc1 : acc0;
    const int gmb = blockM + wm * 32 + mt * 16 + mo;
#pragma unroll
    for (int r = 0; r < 8; ++r) {
      const float y = sigmoidf(acc[r] + bev);
      yenc[(size_t)(gmb + r) * FEATP + gn] = nok ? f32_bf16(y) : (unsigned short)0;
      const float sder = y * (1.f - y);
      jac += nok ? sder * sder * rnv : 0.f;
    }
  }
  for (int off = 16; off; off >>= 1) jac += __shfl_xor(jac, off, 32);
  if (lane == 0) red[w] = jac;
  __syncthreads();
  if (t == 0) {
    float sjac = 0.f;
    for (int i = 0; i < 8; ++i) sjac += red[i];
    jacPart[blockIdx.y * gridDim.x + blockIdx.x] = sjac;
  }
}

// ---------------- jac_reg = sum of 96 block partials (fixed order) ----------
__global__ void cae_jacreduce(const float* __restrict__ part, float* __restrict__ out) {
  if (threadIdx.x == 0) {
    float s = 0.f;
    for (int i = 0; i < 96; ++i) s += part[i];
    out[0] = s;
  }
}

// ---------------- decoder GEMM: y_out = sigmoid(yenc @ wdecb^T + b) ---------
__global__ __launch_bounds__(256) void cae_decoder(
    const unsigned short* __restrict__ yenc, const unsigned short* __restrict__ wb,
    const float* __restrict__ bdec, float* __restrict__ yout) {
  __shared__ __align__(16) unsigned short As[NBUF * TILE];
  __shared__ __align__(16) unsigned short Bs[NBUF * TILE];

  const int t = threadIdx.x;
  const int blockM = blockIdx.y * 64;
  const int blockN = blockIdx.x * 64;

  const int srow = t >> 3;
  const int scol = (t & 7) * 8;
  const unsigned short* pA0 = yenc + (size_t)(blockM + srow) * FEATP + scol;
  const unsigned short* pA1 = pA0 + (size_t)32 * FEATP;
  const unsigned short* pB0 = wb + (size_t)(blockN + srow) * FEATP + scol;
  const unsigned short* pB1 = pB0 + (size_t)32 * FEATP;

  const int lane = t & 31;
  const int w    = t >> 5;
  const int wm   = w >> 2;
  const int wn   = w & 3;
  const int lrow = lane & 15;
  const int kg   = lane >> 4;

  const unsigned short* bfragp  = Bs + (wn * 16 + lrow) * LDT + kg * 8;
  const unsigned short* afragp0 = As + (wm * 32 + lrow) * LDT + kg * 8;
  const unsigned short* afragp1 = afragp0 + 16 * LDT;

  v8f acc0 = {};
  v8f acc1 = {};
  gemm_mainloop<FEATP / BK>(pA0, pA1, pB0, pB1, As, Bs, srow, scol, bfragp, afragp0,
                            afragp1, acc0, acc1);

  const int gn = blockN + wn * 16 + lrow;  // always < 28224 (441*64 exact)
  const float bev = bdec[gn];
  const int mo = kg * 8;
#pragma unroll
  for (int mt = 0; mt < 2; ++mt) {
    const v8f acc = mt ? acc1 : acc0;
    const int gmb = blockM + wm * 32 + mt * 16 + mo;
#pragma unroll
    for (int r = 0; r < 8; ++r) {
      yout[(size_t)(gmb + r) * IN + gn] = sigmoidf(acc[r] + bev);
    }
  }
}

extern "C" void kernel_launch(void* const* d_in, const int* in_sizes, int n_in,
                              void* d_out, int out_size, void* d_ws, size_t ws_size,
                              hipStream_t stream) {
  const float* x    = (const float*)d_in[0];
  const float* Wenc = (const float*)d_in[1];
  const float* benc = (const float*)d_in[2];
  const float* Wdec = (const float*)d_in[3];
  const float* bdec = (const float*)d_in[4];
  float* out = (float*)d_out;  // [256*28224] y_out, then jac_reg scalar

  // ws layout (~178 MB): rn2[1536] | jacPart[128] | yenc bf16[256*1536]
  //   | xb bf16[256*28224] | wencb bf16[1500*28224] | wdecb bf16[28224*1536]
  float* rn2 = (float*)d_ws;
  float* jacPart = rn2 + 1536;
  unsigned short* yenc  = (unsigned short*)(jacPart + 128);
  unsigned short* xb    = yenc + (size_t)BATCH * FEATP;
  unsigned short* wencb = xb + (size_t)BATCH * IN;
  unsigned short* wdecb = wencb + (size_t)FEAT * IN;

  cae_cvt_x<<<(BATCH * IN) / (256 * 8), 256, 0, stream>>>(x, xb);
  cae_prep_wenc<<<FEAT, 256, 0, stream>>>(Wenc, rn2, wencb);
  cae_cvt_wdec<<<IN, 192, 0, stream>>>(Wdec, wdecb);
  cae_encoder<<<dim3(FEATP / 64, BATCH / 64), 256, 0, stream>>>(xb, wencb, benc, rn2,
                                                                yenc, jacPart);
  cae_jacreduce<<<1, 32, 0, stream>>>(jacPart, out + (size_t)BATCH * IN);
  cae_decoder<<<dim3(IN / 64, BATCH / 64), 256, 0, stream>>>(yenc, wdecb, bdec, out);
}
